// mix_model_10385230922264
// MI455X (gfx1250) — matrix operand order repaired and checked by reading, not compiled, not run on hardware
//
#include <hip/hip_runtime.h>
#include <hip/hip_bf16.h>
#include <math.h>

// ---------------------------------------------------------------------------
// Shapes from the reference
// ---------------------------------------------------------------------------
#define B_   64
#define N_   100
#define T_   50
#define DIN  318
#define ROWS (B_ * N_ * T_)     /* 320000 */
#define MTILES (ROWS / 16)      /* 20000  */

typedef __attribute__((ext_vector_type(16))) __bf16 v16bf;
typedef __attribute__((ext_vector_type(8)))  float  v8f;

__device__ __forceinline__ float selu_f(float x) {
    const float scale = 1.0507009873554805f;
    const float alpha = 1.6732632423543772f;
    return scale * (x > 0.f ? x : alpha * (expf(x) - 1.f));
}

// ---------------------------------------------------------------------------
// Dynamic LDS layout for kernel 1
//   sW1: [prec][kt(10)][nt(4)][lane(32)*16] bf16  = 81920 B
//   sW2: [prec][kt(2)][nt(2)][512]          bf16  =  8192 B
//   sW3: [prec][kt(1)][nt(2)][512]          bf16  =  4096 B
//   sH1: 4 waves * 16*64 f32                      = 16384 B
//   sH2: 4 waves * 16*32 f32                      =  8192 B
// ---------------------------------------------------------------------------
#define W1_ELEMS (10 * 4 * 512)
#define W2_ELEMS (2 * 2 * 512)
#define W3_ELEMS (1 * 2 * 512)
#define W1_OFF   0
#define W2_OFF   (W1_OFF + 2 * W1_ELEMS * 2)
#define W3_OFF   (W2_OFF + 2 * W2_ELEMS * 2)
#define H1_OFF   (W3_OFF + 2 * W3_ELEMS * 2)
#define H2_OFF   (H1_OFF + 4 * 16 * 64 * 4)
#define SMEM_BYTES (H2_OFF + 4 * 16 * 32 * 4)

__device__ __forceinline__ v8f wmma3(v16bf ah, v16bf al,
                                     const __bf16* bhp, const __bf16* blp, v8f c) {
    v16bf bh = *(const v16bf*)bhp;
    v16bf bl = *(const v16bf*)blp;
    c = __builtin_amdgcn_wmma_f32_16x16x32_bf16(false, ah, false, bh, (short)0, c, false, false);
    c = __builtin_amdgcn_wmma_f32_16x16x32_bf16(false, ah, false, bl, (short)0, c, false, false);
    c = __builtin_amdgcn_wmma_f32_16x16x32_bf16(false, al, false, bh, (short)0, c, false, false);
    return c;
}

// ---------------------------------------------------------------------------
// Kernel 1: h3 = selu(selu(selu(x@w1+b1)@w2+b2)@w3+b3)  -> (ROWS, 20)
// One wave (32 lanes) per 16-row M-tile, bf16x3 split-precision WMMA.
// ---------------------------------------------------------------------------
__global__ __launch_bounds__(128, 1)
void mlp3_wmma_kernel(const float* __restrict__ x,
                      const float* __restrict__ w1, const float* __restrict__ b1,
                      const float* __restrict__ w2, const float* __restrict__ b2,
                      const float* __restrict__ w3, const float* __restrict__ b3,
                      float* __restrict__ h3out) {
    extern __shared__ char smem[];
    __bf16* sW1 = (__bf16*)(smem + W1_OFF);
    __bf16* sW2 = (__bf16*)(smem + W2_OFF);
    __bf16* sW3 = (__bf16*)(smem + W3_OFF);
    float*  sH1 = (float*)(smem + H1_OFF);
    float*  sH2 = (float*)(smem + H2_OFF);

    // ---- cooperative weight -> hi/lo bf16 fragment conversion --------------
    // B fragments in this code's contiguous k order: lane L holds column n = L&15,
    // 16 consecutive K at kbase = kt*32 + (L<16 ? 0 : 16).
    for (int idx = threadIdx.x; idx < W1_ELEMS; idx += blockDim.x) {
        int i  = idx & 15;
        int ln = (idx >> 4) & 31;
        int nt = (idx >> 9) & 3;
        int kt = idx >> 11;
        int n  = nt * 16 + (ln & 15);
        int k  = kt * 32 + ((ln < 16) ? 0 : 16) + i;
        float v = (k < DIN && n < 50) ? w1[k * 50 + n] : 0.f;
        __bf16 h = (__bf16)v;
        sW1[idx]            = h;
        sW1[W1_ELEMS + idx] = (__bf16)(v - (float)h);
    }
    for (int idx = threadIdx.x; idx < W2_ELEMS; idx += blockDim.x) {
        int i  = idx & 15;
        int ln = (idx >> 4) & 31;
        int nt = (idx >> 9) & 1;
        int kt = idx >> 10;
        int n  = nt * 16 + (ln & 15);
        int k  = kt * 32 + ((ln < 16) ? 0 : 16) + i;
        float v = (k < 50 && n < 20) ? w2[k * 20 + n] : 0.f;
        __bf16 h = (__bf16)v;
        sW2[idx]            = h;
        sW2[W2_ELEMS + idx] = (__bf16)(v - (float)h);
    }
    for (int idx = threadIdx.x; idx < W3_ELEMS; idx += blockDim.x) {
        int i  = idx & 15;
        int ln = (idx >> 4) & 31;
        int nt = (idx >> 9) & 1;
        int n  = nt * 16 + (ln & 15);
        int k  = ((ln < 16) ? 0 : 16) + i;
        float v = (k < 20 && n < 20) ? w3[k * 20 + n] : 0.f;
        __bf16 h = (__bf16)v;
        sW3[idx]            = h;
        sW3[W3_ELEMS + idx] = (__bf16)(v - (float)h);
    }
    __syncthreads();

    const int lane = threadIdx.x & 31;
    const int wave = threadIdx.x >> 5;
    const int nlo  = lane & 15;                 // column within 16-wide tile
    const int koff = (lane < 16) ? 0 : 16;      // A-fragment K offset
    const int moff = (lane < 16) ? 0 : 8;       // C-fragment row offset

    // bias values (n-dependent, loop invariant)
    float b1v[4], b2v[2], b3v[2];
#pragma unroll
    for (int nt = 0; nt < 4; ++nt) { int n = nt * 16 + nlo; b1v[nt] = (n < 50) ? b1[n] : 0.f; }
#pragma unroll
    for (int nt = 0; nt < 2; ++nt) { int n = nt * 16 + nlo; b2v[nt] = (n < 20) ? b2[n] : 0.f; }
#pragma unroll
    for (int nt = 0; nt < 2; ++nt) { int n = nt * 16 + nlo; b3v[nt] = (n < 20) ? b3[n] : 0.f; }

    float* h1 = sH1 + wave * (16 * 64);
    float* h2 = sH2 + wave * (16 * 32);

    for (int tile = blockIdx.x * 4 + wave; tile < MTILES; tile += gridDim.x * 4) {
        const float* xrow = x + ((size_t)tile * 16 + nlo) * DIN;  // row = lane&15

        // ---------------- layer 1: (16x318) @ (318x50) ----------------------
        v8f acc[4];
#pragma unroll
        for (int nt = 0; nt < 4; ++nt)
#pragma unroll
            for (int v = 0; v < 8; ++v) acc[nt][v] = b1v[nt];

#pragma unroll
        for (int kt = 0; kt < 10; ++kt) {
            const int kb = kt * 32 + koff;
            v16bf ah, al;
#pragma unroll
            for (int i = 0; i < 16; ++i) {
                int k = kb + i;
                float v = (kt < 9) ? xrow[k] : ((k < DIN) ? xrow[k] : 0.f);
                __bf16 h = (__bf16)v;
                ah[i] = h;
                al[i] = (__bf16)(v - (float)h);
            }
#pragma unroll
            for (int nt = 0; nt < 4; ++nt) {
                const int fo = ((kt * 4 + nt) * 32 + lane) * 16;
                acc[nt] = wmma3(ah, al, &sW1[fo], &sW1[W1_ELEMS + fo], acc[nt]);
            }
        }
        // SELU + stage h1 tile (16 x 64) in LDS
#pragma unroll
        for (int nt = 0; nt < 4; ++nt)
#pragma unroll
            for (int v = 0; v < 8; ++v)
                h1[(v + moff) * 64 + nt * 16 + nlo] = selu_f(acc[nt][v]);
        asm volatile("s_wait_dscnt 0" ::: "memory");

        // ---------------- layer 2: (16x50) @ (50x20) ------------------------
        v8f acc2[2];
#pragma unroll
        for (int nt = 0; nt < 2; ++nt)
#pragma unroll
            for (int v = 0; v < 8; ++v) acc2[nt][v] = b2v[nt];

#pragma unroll
        for (int kt = 0; kt < 2; ++kt) {
            const int kb = kt * 32 + koff;
            v16bf ah, al;
#pragma unroll
            for (int i = 0; i < 16; ++i) {
                int k = kb + i;
                float v = h1[nlo * 64 + k];
                __bf16 h = (__bf16)v;
                ah[i] = h;
                al[i] = (__bf16)(v - (float)h);
            }
#pragma unroll
            for (int nt = 0; nt < 2; ++nt) {
                const int fo = ((kt * 2 + nt) * 32 + lane) * 16;
                acc2[nt] = wmma3(ah, al, &sW2[fo], &sW2[W2_ELEMS + fo], acc2[nt]);
            }
        }
#pragma unroll
        for (int nt = 0; nt < 2; ++nt)
#pragma unroll
            for (int v = 0; v < 8; ++v)
                h2[(v + moff) * 32 + nt * 16 + nlo] = selu_f(acc2[nt][v]);
        asm volatile("s_wait_dscnt 0" ::: "memory");

        // ---------------- layer 3: (16x20) @ (20x20) ------------------------
        v8f acc3[2];
#pragma unroll
        for (int nt = 0; nt < 2; ++nt)
#pragma unroll
            for (int v = 0; v < 8; ++v) acc3[nt][v] = b3v[nt];
        {
            v16bf ah, al;
#pragma unroll
            for (int i = 0; i < 16; ++i) {
                int k = koff + i;
                float v = h2[nlo * 32 + k];
                __bf16 h = (__bf16)v;
                ah[i] = h;
                al[i] = (__bf16)(v - (float)h);
            }
#pragma unroll
            for (int nt = 0; nt < 2; ++nt) {
                const int fo = (nt * 32 + lane) * 16;
                acc3[nt] = wmma3(ah, al, &sW3[fo], &sW3[W3_ELEMS + fo], acc3[nt]);
            }
        }
        // write h3 (only cols < 20)
#pragma unroll
        for (int nt = 0; nt < 2; ++nt) {
            int n = nt * 16 + nlo;
            if (n < 20) {
#pragma unroll
                for (int v = 0; v < 8; ++v)
                    h3out[((size_t)tile * 16 + v + moff) * 20 + n] = selu_f(acc3[nt][v]);
            }
        }
    }
}

// ---------------------------------------------------------------------------
// Kernel 2: attention pool over T (6400 groups of 50 rows x 20 feats)
// ---------------------------------------------------------------------------
__global__ void attn_pool_t_kernel(const float* __restrict__ h3,
                                   const float* __restrict__ pw1, const float* __restrict__ pb1,
                                   const float* __restrict__ pw2, const float* __restrict__ pb2,
                                   float* __restrict__ out) {
    int g = blockIdx.x;                       // 0 .. B_*N_-1
    const float* r0 = h3 + (size_t)g * T_ * 20;
    __shared__ float sc[T_];
    __shared__ float inv[1];
    int t = threadIdx.x;
    if (t < T_) {
        const float* r = r0 + t * 20;
        float s = pb2[0];
        for (int j = 0; j < 10; ++j) {
            float a = pb1[j];
            for (int f = 0; f < 20; ++f) a += r[f] * pw1[f * 10 + j];
            s += fmaxf(a, 0.f) * pw2[j];
        }
        sc[t] = s;
    }
    __syncthreads();
    if (t == 0) {
        float mx = -3.4e38f;
        for (int i = 0; i < T_; ++i) mx = fmaxf(mx, sc[i]);
        float sm = 0.f;
        for (int i = 0; i < T_; ++i) { float e = expf(sc[i] - mx); sc[i] = e; sm += e; }
        inv[0] = 1.f / sm;
    }
    __syncthreads();
    if (t < 20) {
        float a = 0.f;
        for (int i = 0; i < T_; ++i) a += r0[i * 20 + t] * sc[i];
        out[(size_t)g * 20 + t] = a * inv[0];
    }
}

// ---------------------------------------------------------------------------
// Kernel 3: attention pool over N, L2 normalize, final 20->10->1 MLP
// ---------------------------------------------------------------------------
__global__ void pool_n_final_kernel(const float* __restrict__ rep,      /* (B_, N_, 20) */
                                    const float* __restrict__ pw1, const float* __restrict__ pb1,
                                    const float* __restrict__ pw2, const float* __restrict__ pb2,
                                    const float* __restrict__ mw1, const float* __restrict__ mb1,
                                    const float* __restrict__ mw2, const float* __restrict__ mb2,
                                    float* __restrict__ out) {
    int b = blockIdx.x;
    const float* r0 = rep + (size_t)b * N_ * 20;
    __shared__ float sc[N_];
    __shared__ float hv[20];
    __shared__ float inv[1];
    int t = threadIdx.x;
    if (t < N_) {
        const float* r = r0 + t * 20;
        float s = pb2[0];
        for (int j = 0; j < 10; ++j) {
            float a = pb1[j];
            for (int f = 0; f < 20; ++f) a += r[f] * pw1[f * 10 + j];
            s += fmaxf(a, 0.f) * pw2[j];
        }
        sc[t] = s;
    }
    __syncthreads();
    if (t == 0) {
        float mx = -3.4e38f;
        for (int i = 0; i < N_; ++i) mx = fmaxf(mx, sc[i]);
        float sm = 0.f;
        for (int i = 0; i < N_; ++i) { float e = expf(sc[i] - mx); sc[i] = e; sm += e; }
        inv[0] = 1.f / sm;
    }
    __syncthreads();
    if (t < 20) {
        float a = 0.f;
        for (int i = 0; i < N_; ++i) a += r0[i * 20 + t] * sc[i];
        hv[t] = a * inv[0];
    }
    __syncthreads();
    if (t == 0) {
        float nrm = 0.f;
        for (int f = 0; f < 20; ++f) nrm += hv[f] * hv[f];
        float in2 = 1.f / fmaxf(sqrtf(nrm), 1e-12f);
        float o = mb2[0];
        for (int j = 0; j < 10; ++j) {
            float a = mb1[j];
            for (int f = 0; f < 20; ++f) a += hv[f] * in2 * mw1[f * 10 + j];
            o += selu_f(a) * mw2[j];
        }
        out[b] = o;
    }
}

// ---------------------------------------------------------------------------
// kernel_launch
// ---------------------------------------------------------------------------
extern "C" void kernel_launch(void* const* d_in, const int* in_sizes, int n_in,
                              void* d_out, int out_size, void* d_ws, size_t ws_size,
                              hipStream_t stream) {
    const float* x      = (const float*)d_in[0];
    const float* w1     = (const float*)d_in[1];
    const float* b1     = (const float*)d_in[2];
    const float* w2     = (const float*)d_in[3];
    const float* b2     = (const float*)d_in[4];
    const float* w3     = (const float*)d_in[5];
    const float* b3     = (const float*)d_in[6];
    const float* p2_w1  = (const float*)d_in[7];
    const float* p2_b1  = (const float*)d_in[8];
    const float* p2_w2  = (const float*)d_in[9];
    const float* p2_b2  = (const float*)d_in[10];
    const float* p21_w1 = (const float*)d_in[11];
    const float* p21_b1 = (const float*)d_in[12];
    const float* p21_w2 = (const float*)d_in[13];
    const float* p21_b2 = (const float*)d_in[14];
    const float* m3_w1  = (const float*)d_in[15];
    const float* m3_b1  = (const float*)d_in[16];
    const float* m3_w2  = (const float*)d_in[17];
    const float* m3_b2  = (const float*)d_in[18];

    float* h3 = (float*)d_ws;                       // ROWS * 20 f32
    float* p1 = h3 + (size_t)ROWS * 20;             // B_*N_ * 20 f32

    mlp3_wmma_kernel<<<1250, 128, SMEM_BYTES, stream>>>(x, w1, b1, w2, b2, w3, b3, h3);
    attn_pool_t_kernel<<<B_ * N_, 64, 0, stream>>>(h3, p2_w1, p2_b1, p2_w2, p2_b2, p1);
    pool_n_final_kernel<<<B_, 128, 0, stream>>>(p1, p21_w1, p21_b1, p21_w2, p21_b2,
                                                m3_w1, m3_b1, m3_w2, m3_b2, (float*)d_out);
}
